// BagModel_70119636075014
// MI455X (gfx1250) — compile-verified, matrix operand order checked by reading, not run on hardware
//
#include <hip/hip_runtime.h>
#include <hip/hip_bf16.h>

typedef _Float16 v16h __attribute__((ext_vector_type(16)));
typedef float    v8f  __attribute__((ext_vector_type(8)));

#define BAGS    8192
#define NI      4096
#define THREADS 256
#define PER_TH  (NI / THREADS)   // 16 elements per thread

__global__ __launch_bounds__(THREADS)
void bag_topk_sort_kernel(const float* __restrict__ X,
                          const float* __restrict__ mask,
                          float* __restrict__ probs_out,   // [B]
                          int*   __restrict__ order_out,   // [B,N]
                          int*   __restrict__ k_out)       // [B]
{
    __shared__ unsigned long long sm_keys[NI];   // 32 KB: becomes the sort array
    __shared__ float sh_red[THREADS];
    __shared__ int   sh_k;

    const int row = blockIdx.x;
    const int t   = threadIdx.x;
    const float* Xrow = X    + (size_t)row * NI;
    const float* Mrow = mask + (size_t)row * NI;

    // ---- Phase 1: async global->LDS scatter (CDNA5 ASYNCcnt path). ----
    // X[i] -> high dword of key slot i, mask[i] -> low dword of key slot i.
    // Coalesced 4B/lane global reads; LDS writes are an 8B-stride scatter.
    for (int c = 0; c < PER_TH; ++c) {
        int i = c * THREADS + t;
        unsigned lo_lds = (unsigned)(size_t)&sm_keys[i];
        unsigned hi_lds = lo_lds + 4u;
        unsigned long long gx = (unsigned long long)(size_t)(Xrow + i);
        unsigned long long gm = (unsigned long long)(size_t)(Mrow + i);
        asm volatile("global_load_async_to_lds_b32 %0, %1, off"
                     :: "v"(hi_lds), "v"(gx) : "memory");
        asm volatile("global_load_async_to_lds_b32 %0, %1, off"
                     :: "v"(lo_lds), "v"(gm) : "memory");
    }
    asm volatile("s_wait_asynccnt 0" ::: "memory");
    __syncthreads();

    // ---- Phase 2 (wave 0, EXEC all ones): row-sum of mask on the matrix pipe.
    // B = all-ones, so D[m][n] = sum_k A[m][k]; total = sum over all A elems,
    // invariant to lane<->(M,K) placement. mask in {0,1} is exact in f16.
    if (t < 32) {
        const unsigned* sm32 = (const unsigned*)sm_keys;
        v16h ones;
        for (int c = 0; c < 16; ++c) ones[c] = (_Float16)1.0f;
        v8f acc = {};
        for (int it = 0; it < NI / 512; ++it) {
            v16h a;
            for (int c = 0; c < 16; ++c) {
                int e = it * 512 + t + 32 * c;              // disjoint cover of 512 elems
                a[c] = (_Float16)__uint_as_float(sm32[2 * e]); // low dword = mask value
            }
            acc = __builtin_amdgcn_wmma_f32_16x16x32_f16(
                      false, a, false, ones, (short)0, acc, false, false);
        }
        float s = acc[0] + acc[1] + acc[2] + acc[3] + acc[4] + acc[5] + acc[6] + acc[7];
        s += __shfl_xor(s, 16, 32);                         // fold the two lane halves
        if (t == 0) {
            int kk = (int)floorf(0.2f * s);                 // KEY_INSTANCE_RATIO
            if (kk < 1) kk = 1;
            k_out[row] = kk;
            sh_k = kk;
        }
    }
    __syncthreads();

    // ---- Phase 3: build stable-descending sort keys in place. ----
    // key = (bits(X*mask) << 32) | ~index ; vals >= 0 so f32 bits order
    // monotonically, ~index breaks ties toward the smaller index (stable).
    for (int c = 0; c < PER_TH; ++c) {
        int i = c * THREADS + t;
        unsigned long long w = sm_keys[i];
        float m   = __uint_as_float((unsigned)w);
        float x   = __uint_as_float((unsigned)(w >> 32));
        float val = x * m;
        sm_keys[i] = ((unsigned long long)__float_as_uint(val) << 32)
                   | (unsigned)(~i);
    }
    __syncthreads();

    // ---- Phase 4: descending bitonic sort of 4096 u64 keys in LDS. ----
    // 78 substages, 2048 disjoint pairs each, 8 pairs per thread,
    // conflict-free b64 DS accesses (consecutive lanes 8B apart).
    for (int k2 = 2; k2 <= NI; k2 <<= 1) {
        for (int j = k2 >> 1; j >= 1; j >>= 1) {
            for (int c = 0; c < NI / 2 / THREADS; ++c) {
                int p = c * THREADS + t;
                int i = ((p & ~(j - 1)) << 1) | (p & (j - 1));
                int q = i | j;
                unsigned long long a = sm_keys[i];
                unsigned long long b = sm_keys[q];
                bool sw = ((i & k2) == 0) ? (a < b) : (a > b);   // descending net
                if (sw) { sm_keys[i] = b; sm_keys[q] = a; }
            }
            __syncthreads();
        }
    }

    // ---- Phase 5: mean of top-k (tree reduction). ----
    int kk = sh_k;
    float part = 0.0f;
    for (int c = 0; c < PER_TH; ++c) {
        int i = c * THREADS + t;
        if (i < kk) part += __uint_as_float((unsigned)(sm_keys[i] >> 32));
    }
    sh_red[t] = part;
    __syncthreads();
    for (int sft = THREADS / 2; sft > 0; sft >>= 1) {
        if (t < sft) sh_red[t] += sh_red[t + sft];
        __syncthreads();
    }
    if (t == 0) probs_out[row] = sh_red[0] / (float)kk;

    // ---- Phase 6: write the full descending ordering (coalesced b32). ----
    int* orow = order_out + (size_t)row * NI;
    for (int c = 0; c < PER_TH; ++c) {
        int i = c * THREADS + t;
        orow[i] = (int)~((unsigned)sm_keys[i]);   // recover original index
    }
}

extern "C" void kernel_launch(void* const* d_in, const int* in_sizes, int n_in,
                              void* d_out, int out_size, void* d_ws, size_t ws_size,
                              hipStream_t stream) {
    (void)in_sizes; (void)n_in; (void)out_size; (void)d_ws; (void)ws_size;
    const float* X    = (const float*)d_in[0];
    const float* mask = (const float*)d_in[1];
    // d_out layout (reference return order): probs[B] f32 | order[B*N] i32 | k[B] i32
    float* out   = (float*)d_out;
    float* probs = out;
    int*   order = (int*)(out + BAGS);
    int*   kv    = (int*)(out + BAGS + (size_t)BAGS * NI);
    bag_topk_sort_kernel<<<BAGS, THREADS, 0, stream>>>(X, mask, probs, order, kv);
}
